// BLT_encoder_32203664786067
// MI455X (gfx1250) — matrix operand order repaired and checked by reading, not compiled, not run on hardware
//
#include <hip/hip_runtime.h>
#include <hip/hip_bf16.h>
#include <math.h>

// ---------------- problem constants ----------------
constexpr int Bq = 4, Tq = 1024, Cq = 768, Hq = 12, Lq = 6, Vq = 32000;
constexpr int DHq = Cq / Hq;          // 64
constexpr int Mq = Bq * Tq;           // 4096 rows

// ---------------- WMMA / TDM types ----------------
typedef __bf16        v16bf __attribute__((ext_vector_type(16)));
typedef float         v8f   __attribute__((ext_vector_type(8)));
typedef unsigned int  u32x4 __attribute__((ext_vector_type(4)));
typedef int           i32x8 __attribute__((ext_vector_type(8)));
typedef int           i32x4 __attribute__((ext_vector_type(4)));

union Frag16 { uint4 q[2]; v16bf v; };   // 32 bytes = one 16x32 bf16 A/B fragment per lane

__device__ inline v8f zero_v8f() {
    v8f z = {0.f, 0.f, 0.f, 0.f, 0.f, 0.f, 0.f, 0.f};
    return z;
}

// Issue one TDM 2D tile load: global (rows x 64 bf16, row stride K elems) -> LDS,
// LDS rows padded by 8 bf16 (4 DWORDs) every 64 bf16 (32 DWORDs).
// D# layout per CDNA5 ISA.
// (6-arg builtin form for the clang-23 / therock-10.0 toolchain.)
__device__ inline void tdm_load_tile(const __bf16* gptr, unsigned lds_addr, int K, int rows)
{
    unsigned long long ga = (unsigned long long)(size_t)gptr;
    u32x4 g0;
    g0[0] = 1u;                                   // count=1, user descriptor
    g0[1] = lds_addr;                             // lds_addr (bytes)
    g0[2] = (unsigned)ga;                         // global_addr[31:0]
    g0[3] = (unsigned)(ga >> 32) | (2u << 30);    // global_addr[56:32] | type=2 (image)
    i32x8 g1;
    g1[0] = (1 << 16)     // data_size = 1 -> 2 bytes
          | (1 << 20)     // pad_enable
          | (4 << 22)     // pad_interval code 4 -> every 32 DWORDs (64 bf16)
          | (3 << 25);    // pad_amount  code 3 -> 4 DWORDs (8 bf16)
    g1[1] = (K & 0xFFFF) << 16;                            // tensor_dim0[15:0]
    g1[2] = ((unsigned)K >> 16) | ((rows & 0xFFFF) << 16); // tensor_dim0[31:16] | tensor_dim1[15:0]
    g1[3] = (((unsigned)rows >> 16) & 0xFFFF) | (64u << 16); // tensor_dim1[31:16] | tile_dim0=64
    g1[4] = rows & 0xFFFF;                                 // tile_dim1 = rows (tile_dim2=0)
    g1[5] = K;                                             // tensor_dim0_stride[31:0] (elems)
    g1[6] = 0;                                             // stride hi / dim1_stride lo
    g1[7] = 0;
    i32x4 gz4 = {0, 0, 0, 0};                              // groups 2/3 unused (2D tensor)
    i32x8 gz8 = {0, 0, 0, 0, 0, 0, 0, 0};
    __builtin_amdgcn_tensor_load_to_lds(g0, g1, gz4, gz4, gz8, 0);
}

// ===========================================================================
// 1) embedding: x[b,t,:] = tok_emb[idx[b,t],:] + pos_emb[t,:]   (fp32)
// ===========================================================================
__global__ __launch_bounds__(256) void embed_kernel(
    const int* __restrict__ idx, const float* __restrict__ tok,
    const float* __restrict__ pos, float* __restrict__ x)
{
    int row = blockIdx.x;              // 0..B*T-1
    int t   = row % Tq;
    int tk  = idx[row];
    const float* tr = tok + (size_t)tk * Cq;
    const float* pr = pos + (size_t)t  * Cq;
    float* xr = x + (size_t)row * Cq;
    for (int c = threadIdx.x; c < Cq; c += 256)
        xr[c] = tr[c] + pr[c];
}

// ===========================================================================
// 2) LayerNorm (fp32 in -> bf16 out), one block per row, biased variance
// ===========================================================================
__global__ __launch_bounds__(256) void ln_kernel(
    const float* __restrict__ x, const float* __restrict__ g,
    const float* __restrict__ b, __bf16* __restrict__ out)
{
    __shared__ float red[256];
    int row = blockIdx.x, tid = threadIdx.x;
    const float* xr = x + (size_t)row * Cq;

    float s = 0.f;
    for (int c = tid; c < Cq; c += 256) s += xr[c];
    red[tid] = s; __syncthreads();
    for (int st = 128; st > 0; st >>= 1) {
        if (tid < st) red[tid] += red[tid + st];
        __syncthreads();
    }
    float mean = red[0] / (float)Cq;
    __syncthreads();

    float v = 0.f;
    for (int c = tid; c < Cq; c += 256) { float d = xr[c] - mean; v += d * d; }
    red[tid] = v; __syncthreads();
    for (int st = 128; st > 0; st >>= 1) {
        if (tid < st) red[tid] += red[tid + st];
        __syncthreads();
    }
    float inv = rsqrtf(red[0] / (float)Cq + 1e-5f);

    __bf16* orow = out + (size_t)row * Cq;
    for (int c = tid; c < Cq; c += 256)
        orow[c] = (__bf16)((xr[c] - mean) * inv * g[c] + b[c]);
}

// ===========================================================================
// 3) fp32 -> bf16 convert WITH transpose: W[K][N] -> Wt[N][K]
// ===========================================================================
__global__ __launch_bounds__(256) void cvtT_kernel(
    const float* __restrict__ W, __bf16* __restrict__ Wt, int K, int N)
{
    __shared__ float t[32][33];
    int k0 = blockIdx.y * 32, n0 = blockIdx.x * 32;
    int tx = threadIdx.x & 31, ty = threadIdx.x >> 5;   // 32 x 8
    #pragma unroll
    for (int i = 0; i < 32; i += 8)
        t[ty + i][tx] = W[(size_t)(k0 + ty + i) * N + n0 + tx];
    __syncthreads();
    #pragma unroll
    for (int i = 0; i < 32; i += 8)
        Wt[(size_t)(n0 + ty + i) * K + k0 + tx] = (__bf16)t[tx][ty + i];
}

// ===========================================================================
// 4) WMMA GEMM with double-buffered TDM staging:
//    D[M,N] = A[M,K](bf16,rm) @ W (pre-transposed Wt[N][K]) + bias
//    Block tile 128x128, BK=64; A/B tiles DMA'd into LDS by the Tensor Data
//    Mover; tile k+1's DMA is issued before computing tile k, waited with
//    s_wait_tensorcnt(1) (tensor ops complete in order per wave).
//    8 waves (2Mx4N), wave tile 64x32: 16 wmma per wave per staging round.
// ===========================================================================
__global__ __launch_bounds__(256) void gemm_bf16(
    const __bf16* __restrict__ A, const __bf16* __restrict__ Bt,
    const float* __restrict__ bias, const float* __restrict__ res,
    float* __restrict__ outF, __bf16* __restrict__ outH,
    int M, int N, int K, int gelu, int transV)
{
    constexpr int BK = 64;                                   // row stride 72 bf16 = 144B
    __shared__ __align__(16) __bf16 As[2][128][BK + 8];      // double-buffered [m][k]
    __shared__ __align__(16) __bf16 Bs[2][128][BK + 8];      // double-buffered [n][k]

    int tid  = threadIdx.x;
    int wave = tid >> 5, lane = tid & 31;
    int hf   = lane >> 4, l16 = lane & 15;
    int wm   = wave >> 2, wn = wave & 3;                  // 2 x 4 waves
    int m0   = blockIdx.y * 128, n0 = blockIdx.x * 128;

    int wid = __builtin_amdgcn_readfirstlane(wave);       // scalar wave id
    const __bf16* aBase = A  + (size_t)m0 * K;
    const __bf16* bBase = Bt + (size_t)n0 * K;

    v8f acc[4][2];
    #pragma unroll
    for (int mt = 0; mt < 4; ++mt)
        #pragma unroll
        for (int nt = 0; nt < 2; ++nt) acc[mt][nt] = zero_v8f();

    // prologue: DMA tile 0 into buffer 0
    if (wid == 0)      tdm_load_tile(aBase, (unsigned)(size_t)(void*)&As[0][0][0], K, 128);
    else if (wid == 1) tdm_load_tile(bBase, (unsigned)(size_t)(void*)&Bs[0][0][0], K, 128);

    int nk = K / BK;
    for (int ki = 0; ki < nk; ++ki) {
        int buf = ki & 1;
        if (ki + 1 < nk) {
            // issue next tile's DMA into the other buffer, then wait for current
            if (wid == 0)
                tdm_load_tile(aBase + (size_t)(ki + 1) * BK,
                              (unsigned)(size_t)(void*)&As[buf ^ 1][0][0], K, 128);
            else if (wid == 1)
                tdm_load_tile(bBase + (size_t)(ki + 1) * BK,
                              (unsigned)(size_t)(void*)&Bs[buf ^ 1][0][0], K, 128);
            __builtin_amdgcn_s_wait_tensorcnt(1);         // in-order: current tile landed
        } else {
            __builtin_amdgcn_s_wait_tensorcnt(0);
        }
        __syncthreads();

        #pragma unroll
        for (int kf = 0; kf < 2; ++kf) {                  // two 32-wide K slices
            Frag16 a[4], b[2];
            #pragma unroll
            for (int mt = 0; mt < 4; ++mt) {
                int r = wm * 64 + mt * 16 + l16;
                a[mt].q[0] = *(const uint4*)(&As[buf][r][kf * 32 + 8 * hf]);
                a[mt].q[1] = *(const uint4*)(&As[buf][r][kf * 32 + 16 + 8 * hf]);
            }
            #pragma unroll
            for (int nt = 0; nt < 2; ++nt) {
                int c = wn * 32 + nt * 16 + l16;
                b[nt].q[0] = *(const uint4*)(&Bs[buf][c][kf * 32 + 8 * hf]);
                b[nt].q[1] = *(const uint4*)(&Bs[buf][c][kf * 32 + 16 + 8 * hf]);
            }
            #pragma unroll
            for (int mt = 0; mt < 4; ++mt)
                #pragma unroll
                for (int nt = 0; nt < 2; ++nt)
                    acc[mt][nt] = __builtin_amdgcn_wmma_f32_16x16x32_bf16(
                        false, a[mt].v, false, b[nt].v, (short)0, acc[mt][nt], false, false);
        }
        __syncthreads();   // all waves done reading buf before it is re-DMA'd (next iter)
    }

    // epilogue: C/D layout -> row m = r + 8*hf, col n = l16
    #pragma unroll
    for (int mt = 0; mt < 4; ++mt) {
        #pragma unroll
        for (int nt = 0; nt < 2; ++nt) {
            int n = n0 + wn * 32 + nt * 16 + l16;
            float bb = bias ? bias[n] : 0.f;
            #pragma unroll
            for (int r = 0; r < 8; ++r) {
                int m = m0 + wm * 64 + mt * 16 + r + 8 * hf;
                float vv = acc[mt][nt][r] + bb;
                if (gelu) vv = 0.5f * vv * (1.f + erff(vv * 0.7071067811865476f));
                if (res)  vv += res[(size_t)m * N + n];
                if (outF) outF[(size_t)m * N + n] = vv;
                if (outH) {
                    size_t oi;
                    if (transV) {                         // [b][h][dh][t]
                        int bb2 = m / Tq, t = m % Tq;
                        int hh = n / DHq, dd = n % DHq;
                        oi = (((size_t)bb2 * Hq + hh) * DHq + dd) * Tq + t;
                    } else {
                        oi = (size_t)m * N + n;
                    }
                    outH[oi] = (__bf16)vv;
                }
            }
        }
    }
}

// ===========================================================================
// 5) Flash attention, WMMA for S=Q*K^T and O=P*V, online softmax.
//    grid = (T/128, B*H); 8 waves, each owns a 16-query tile.
//    V comes in pre-transposed as [b][h][dh][t].
// ===========================================================================
__global__ __launch_bounds__(256) void attn_kernel(
    const __bf16* __restrict__ Q, const __bf16* __restrict__ Kb,
    const __bf16* __restrict__ Vt, const int* __restrict__ idx,
    __bf16* __restrict__ Y)
{
    __shared__ __align__(16) __bf16 Ks[32][64 + 8];     // [key][dh]  stride 144B
    __shared__ __align__(16) __bf16 Vts[64][32 + 8];    // [dh][key]  stride 80B
    __shared__ __align__(16) __bf16 Ps[8][16][32 + 8];  // per-wave P scratch
    __shared__ float msk[32];

    int tid  = threadIdx.x, wave = tid >> 5, lane = tid & 31;
    int hf   = lane >> 4, l16 = lane & 15;
    int bh   = blockIdx.y;
    int b    = bh / Hq, h = bh % Hq;
    int q0   = blockIdx.x * 128 + wave * 16;

    // Q fragments for this wave's 16 queries x 64 dh (two 16x32 A-fragments)
    Frag16 qf[2];
    #pragma unroll
    for (int f = 0; f < 2; ++f) {
        const __bf16* qp = Q + (size_t)(b * Tq + q0 + l16) * Cq + h * DHq + f * 32;
        qf[f].q[0] = *(const uint4*)(qp + 8 * hf);
        qf[f].q[1] = *(const uint4*)(qp + 16 + 8 * hf);
    }

    v8f o[4]; o[0] = zero_v8f(); o[1] = zero_v8f(); o[2] = zero_v8f(); o[3] = zero_v8f();
    float m8[8], l8[8];
    #pragma unroll
    for (int r = 0; r < 8; ++r) { m8[r] = -1e30f; l8[r] = 0.f; }

    const __bf16* vbase = Vt + ((size_t)(b * Hq + h) * DHq) * Tq;

    for (int kt = 0; kt < Tq; kt += 32) {
        // ---- cooperative staging: K row-major, V already transposed ----
        {
            int key = tid >> 3, c = (tid & 7) * 8;      // K tile: [key][dh]
            uint4 dk = *(const uint4*)(Kb + (size_t)(b * Tq + kt + key) * Cq + h * DHq + c);
            *(uint4*)(&Ks[key][c]) = dk;
            if (kt + 32 < Tq)
                __builtin_prefetch(Kb + (size_t)(b * Tq + kt + 32 + key) * Cq + h * DHq + c, 0, 1);
        }
        {
            int dh = tid >> 2, c = (tid & 3) * 8;       // V tile: [dh][key], contiguous rows
            uint4 dv = *(const uint4*)(vbase + (size_t)dh * Tq + kt + c);
            *(uint4*)(&Vts[dh][c]) = dv;
        }
        if (tid < 32) msk[tid] = (idx[b * Tq + kt + tid] != Vq - 1) ? 0.f : -1e30f;
        __syncthreads();

        // ---- S = Q @ K^T  (two 16-key subtiles, K-dim = 64 = 2 wmma) ----
        v8f s[2];
        #pragma unroll
        for (int nt = 0; nt < 2; ++nt) {
            v8f sa = zero_v8f();
            int key = nt * 16 + l16;
            #pragma unroll
            for (int f = 0; f < 2; ++f) {
                Frag16 kf;   // B fragment of K^T: elem e<8 -> K[key][f*32 + 8*hf + e], e>=8 -> K[key][f*32 + 16 + 8*hf + e-8]
                kf.q[0] = *(const uint4*)(&Ks[key][f * 32 + 8 * hf]);
                kf.q[1] = *(const uint4*)(&Ks[key][f * 32 + 16 + 8 * hf]);
                sa = __builtin_amdgcn_wmma_f32_16x16x32_bf16(
                    false, qf[f].v, false, kf.v, (short)0, sa, false, false);
            }
            float mk = msk[key];
            #pragma unroll
            for (int r = 0; r < 8; ++r) sa[r] = sa[r] * 0.125f + mk;  // 1/sqrt(64)
            s[nt] = sa;
        }

        // ---- online softmax (rows live across the 16 lanes of each half) ----
        #pragma unroll
        for (int r = 0; r < 8; ++r) {
            float tmax = fmaxf(s[0][r], s[1][r]);
            #pragma unroll
            for (int d = 1; d < 16; d <<= 1) tmax = fmaxf(tmax, __shfl_xor(tmax, d, 32));
            float mnew = fmaxf(m8[r], tmax);
            float sc = __expf(m8[r] - mnew);
            float p0 = __expf(s[0][r] - mnew);
            float p1 = __expf(s[1][r] - mnew);
            float rs = p0 + p1;
            #pragma unroll
            for (int d = 1; d < 16; d <<= 1) rs += __shfl_xor(rs, d, 32);
            l8[r] = l8[r] * sc + rs;
            m8[r] = mnew;
            o[0][r] *= sc; o[1][r] *= sc; o[2][r] *= sc; o[3][r] *= sc;
            // stash P (bf16) in D-layout position: row r+8*hf, cols l16 / 16+l16
            Ps[wave][r + 8 * hf][l16]      = (__bf16)p0;
            Ps[wave][r + 8 * hf][16 + l16] = (__bf16)p1;
        }
        asm volatile("" ::: "memory");   // intra-wave LDS ops are in-order; block reordering

        // reload P as a 16x32 A-fragment
        Frag16 pf;
        pf.q[0] = *(const uint4*)(&Ps[wave][l16][8 * hf]);
        pf.q[1] = *(const uint4*)(&Ps[wave][l16][16 + 8 * hf]);

        // ---- O += P @ V  (V as B: elem e<8 -> Vts[dh][8*hf+e], e>=8 -> Vts[dh][16+8*hf+e-8]) ----
        #pragma unroll
        for (int dt = 0; dt < 4; ++dt) {
            Frag16 vf;
            int dh = dt * 16 + l16;
            vf.q[0] = *(const uint4*)(&Vts[dh][8 * hf]);
            vf.q[1] = *(const uint4*)(&Vts[dh][16 + 8 * hf]);
            o[dt] = __builtin_amdgcn_wmma_f32_16x16x32_bf16(
                false, pf.v, false, vf.v, (short)0, o[dt], false, false);
        }
        __syncthreads();
    }

    // ---- write Y = O / l ----
    #pragma unroll
    for (int dt = 0; dt < 4; ++dt) {
        #pragma unroll
        for (int r = 0; r < 8; ++r) {
            int m = q0 + r + 8 * hf;
            float inv = 1.f / l8[r];
            Y[(size_t)(b * Tq + m) * Cq + h * DHq + dt * 16 + l16] = (__bf16)(o[dt][r] * inv);
        }
    }
}

// ===========================================================================
// 6) log-softmax + argmax over V per row, in place on the logits region
// ===========================================================================
__global__ __launch_bounds__(256) void lsm_kernel(
    float* __restrict__ logits, float* __restrict__ outIdx, float* __restrict__ outQ)
{
    __shared__ float rv[256];
    __shared__ int   ri[256];
    int row = blockIdx.x, tid = threadIdx.x;
    float* lr = logits + (size_t)row * Vq;

    float mx = -1e30f; int ai = 0;
    for (int c = tid; c < Vq; c += 256) {
        float v = lr[c];
        if (v > mx) { mx = v; ai = c; }
    }
    rv[tid] = mx; ri[tid] = ai; __syncthreads();
    for (int st = 128; st > 0; st >>= 1) {
        if (tid < st) {
            float a = rv[tid], b = rv[tid + st];
            if (b > a || (b == a && ri[tid + st] < ri[tid])) { rv[tid] = b; ri[tid] = ri[tid + st]; }
        }
        __syncthreads();
    }
    mx = rv[0]; ai = ri[0];
    __syncthreads();

    float se = 0.f;
    for (int c = tid; c < Vq; c += 256) se += __expf(lr[c] - mx);
    rv[tid] = se; __syncthreads();
    for (int st = 128; st > 0; st >>= 1) {
        if (tid < st) rv[tid] += rv[tid + st];
        __syncthreads();
    }
    se = rv[0];
    float ls = logf(se);

    for (int c = tid; c < Vq; c += 256) lr[c] = lr[c] - mx - ls;
    if (tid == 0) {
        outIdx[row] = (float)ai;     // seq_new (argmax) as float
        outQ[row]   = 1.f / se;      // seq_new_q = exp(mx-mx)/sumexp
    }
}

// ===========================================================================
// host orchestration
// ===========================================================================
extern "C" void kernel_launch(void* const* d_in, const int* in_sizes, int n_in,
                              void* d_out, int out_size, void* d_ws, size_t ws_size,
                              hipStream_t stream) {
    (void)in_sizes; (void)n_in; (void)out_size; (void)ws_size;

    const int*   idx   = (const int*)  d_in[0];
    const float* tok   = (const float*)d_in[1];
    const float* pos   = (const float*)d_in[2];
    const float* Wq    = (const float*)d_in[3];
    const float* bq    = (const float*)d_in[4];
    const float* Wk    = (const float*)d_in[5];
    const float* bk    = (const float*)d_in[6];
    const float* Wv    = (const float*)d_in[7];
    const float* bv    = (const float*)d_in[8];
    const float* Wo    = (const float*)d_in[9];
    const float* bo    = (const float*)d_in[10];
    const float* ln1g  = (const float*)d_in[11];
    const float* ln1b  = (const float*)d_in[12];
    const float* ln2g  = (const float*)d_in[13];
    const float* ln2b  = (const float*)d_in[14];
    const float* W1    = (const float*)d_in[15];
    const float* b1    = (const float*)d_in[16];
    const float* W2    = (const float*)d_in[17];
    const float* b2    = (const float*)d_in[18];
    const float* lnfg  = (const float*)d_in[19];
    const float* lnfb  = (const float*)d_in[20];
    const float* Whead = (const float*)d_in[21];

    // workspace carve-up
    char* wsb = (char*)d_ws;
    size_t off = 0;
    auto carve = [&](size_t bytes) -> void* {
        void* p = wsb + off;
        off = (off + bytes + 255) & ~(size_t)255;
        return p;
    };
    float*  x0  = (float*) carve((size_t)Mq * Cq * 4);
    float*  x1  = (float*) carve((size_t)Mq * Cq * 4);
    __bf16* hB  = (__bf16*)carve((size_t)Mq * Cq * 2);
    __bf16* qB  = (__bf16*)carve((size_t)Mq * Cq * 2);
    __bf16* kB  = (__bf16*)carve((size_t)Mq * Cq * 2);
    __bf16* vB  = (__bf16*)carve((size_t)Mq * Cq * 2);   // [b][h][dh][t]
    __bf16* yB  = (__bf16*)carve((size_t)Mq * Cq * 2);
    __bf16* mid = (__bf16*)carve((size_t)Mq * 4 * Cq * 2);
    __bf16* wb  = (__bf16*)carve((size_t)Cq * Vq * 2);   // largest weight (Whead), transposed

    float* outF   = (float*)d_out;
    float* seqNew = outF;
    float* seqQ   = outF + Mq;
    float* logits = outF + 2 * (size_t)Mq;

    auto cvtT = [&](const float* src, __bf16* dst, int K, int N) {
        dim3 g(N / 32, K / 32);
        cvtT_kernel<<<g, 256, 0, stream>>>(src, dst, K, N);
    };
    auto gemm = [&](const __bf16* A, const __bf16* Bt, const float* bias,
                    const float* res, float* oF, __bf16* oH,
                    int M, int N, int K, int gelu, int transV) {
        dim3 g(N / 128, M / 128);
        gemm_bf16<<<g, 256, 0, stream>>>(A, Bt, bias, res, oF, oH, M, N, K, gelu, transV);
    };

    // x = tok_emb[idx] + pos_emb
    embed_kernel<<<Mq, 256, 0, stream>>>(idx, tok, pos, x0);

    float* xc = x0;   // current residual stream
    float* xn = x1;

    for (int l = 0; l < Lq; ++l) {
        const size_t wcc = (size_t)Cq * Cq;
        // h = LN1(x)
        ln_kernel<<<Mq, 256, 0, stream>>>(xc, ln1g + l * Cq, ln1b + l * Cq, hB);
        // q,k,v = h @ W{q,k,v} + b   (V output transposed for attention)
        cvtT(Wq + l * wcc, wb, Cq, Cq);
        gemm(hB, wb, bq + l * Cq, nullptr, nullptr, qB, Mq, Cq, Cq, 0, 0);
        cvtT(Wk + l * wcc, wb, Cq, Cq);
        gemm(hB, wb, bk + l * Cq, nullptr, nullptr, kB, Mq, Cq, Cq, 0, 0);
        cvtT(Wv + l * wcc, wb, Cq, Cq);
        gemm(hB, wb, bv + l * Cq, nullptr, nullptr, vB, Mq, Cq, Cq, 0, 1);
        // y = attention(q,k,v) with pad-key mask
        attn_kernel<<<dim3(Tq / 128, Bq * Hq), 256, 0, stream>>>(qB, kB, vB, idx, yB);
        // x = x + y @ Wo + bo
        cvtT(Wo + l * wcc, wb, Cq, Cq);
        gemm(yB, wb, bo + l * Cq, xc, xn, nullptr, Mq, Cq, Cq, 0, 0);
        // h = LN2(x)
        ln_kernel<<<Mq, 256, 0, stream>>>(xn, ln2g + l * Cq, ln2b + l * Cq, hB);
        // mid = gelu(h @ W1 + b1)
        cvtT(W1 + (size_t)l * Cq * 4 * Cq, wb, Cq, 4 * Cq);
        gemm(hB, wb, b1 + (size_t)l * 4 * Cq, nullptr, nullptr, mid, Mq, 4 * Cq, Cq, 1, 0);
        // x = x + mid @ W2 + b2
        cvtT(W2 + (size_t)l * 4 * Cq * Cq, wb, 4 * Cq, Cq);
        gemm(mid, wb, b2 + l * Cq, xn, xc, nullptr, Mq, Cq, 4 * Cq, 0, 0);
        // residual stream ends the layer back in xc
    }

    // final LN + head
    ln_kernel<<<Mq, 256, 0, stream>>>(xc, lnfg, lnfb, hB);
    cvtT(Whead, wb, Cq, Vq);
    gemm(hB, wb, nullptr, nullptr, logits, nullptr, Mq, Vq, Cq, 0, 0);

    // log-softmax + argmax, in place on logits
    lsm_kernel<<<Mq, 256, 0, stream>>>(logits, seqNew, seqQ);
}
